// CrossMambaFusionBlock_77275051589816
// MI455X (gfx1250) — matrix operand order repaired and checked by reading, not compiled, not run on hardware
//
#include <hip/hip_runtime.h>
#include <math.h>

typedef _Float16 v16h __attribute__((ext_vector_type(16)));
typedef float    v8f  __attribute__((ext_vector_type(8)));

#define BB   4
#define HH   64
#define WW_  64
#define CC   96
#define DIN  192
#define NST  4
#define RNK  6
#define KDIR 4
#define LL   (HH * WW_)          // 4096
#define MTOT (BB * LL)           // 16384
#define SCH  128                 // scan LDS chunk (steps per stage)

// Map scan-sequence index p (for direction k) to position in (h*W + w) space.
// k=0: row-major; k=1: transposed (q = w*H + h); k=2: reversed; k=3: transposed reversed.
__device__ __forceinline__ int scan_pos(int k, int p) {
    switch (k & 3) {
        case 0: return p;
        case 1: { int w = p / HH, h = p % HH; return h * WW_ + w; }
        case 2: return LL - 1 - p;
        default: { int q = LL - 1 - p; int w = q / HH, h = q % HH; return h * WW_ + w; }
    }
}

// ---------------- K1: in_proj GEMM, Y[M,N] = X[M,K] * W[N,K]^T, f16 WMMA ----------------
__global__ void k_gemm_in(const float* __restrict__ X0, const float* __restrict__ X1,
                          const float* __restrict__ W0, const float* __restrict__ W1,
                          float* __restrict__ Y0, float* __restrict__ Y1,
                          int M, int N, int Kd)
{
    const float* X  = blockIdx.z ? X1 : X0;
    const float* Wt = blockIdx.z ? W1 : W0;
    float*       Y  = blockIdx.z ? Y1 : Y0;

    int wave = threadIdx.x >> 5;
    int lane = threadIdx.x & 31;
    int tiles = (M >> 4) * (N >> 4);
    int tile  = blockIdx.x * (blockDim.x >> 5) + wave;
    if (tile >= tiles) return;                 // wave-uniform, EXEC stays all-ones below
    int nt = tile % (N >> 4);
    int mt = tile / (N >> 4);

    int row  = lane & 15;       // A: M row / B: N column / D: N column
    int half = lane >> 4;
    int kb8  = half * 8;        // A-matrix K base
    int kb16 = half * 16;       // unused; B also uses kb8

    v8f acc = {0.f, 0.f, 0.f, 0.f, 0.f, 0.f, 0.f, 0.f};
    for (int kk = 0; kk < Kd; kk += 32) {
        v16h a, bm;
        const float* ap = X + (size_t)(mt * 16 + row) * Kd + kk + kb8;
        #pragma unroll
        for (int e = 0; e < 8; ++e) { a[e] = (_Float16)ap[e]; a[8 + e] = (_Float16)ap[16 + e]; }
        const float* bp = Wt + (size_t)(nt * 16 + row) * Kd + kk + kb8;
        #pragma unroll
        for (int e = 0; e < 8; ++e) { bm[e] = (_Float16)bp[e]; bm[8 + e] = (_Float16)bp[16 + e]; }
        acc = __builtin_amdgcn_wmma_f32_16x16x32_f16(false, a, false, bm, (short)0, acc, false, false);
    }
    float* yp = Y + (size_t)(mt * 16 + half * 8) * N + nt * 16 + row;
    #pragma unroll
    for (int v = 0; v < 8; ++v) yp[(size_t)v * N] = acc[v];
}

// ---------------- K2: depthwise 3x3 conv + bias + SiLU; (b,l,d) -> (b,d,l) ----------------
__global__ void k_dwconv(const float* __restrict__ P0, const float* __restrict__ P1,
                         const float* __restrict__ Wc0, const float* __restrict__ Wc1,
                         const float* __restrict__ Bc0, const float* __restrict__ Bc1,
                         float* __restrict__ XC0, float* __restrict__ XC1)
{
    const float* P  = blockIdx.z ? P1 : P0;
    const float* Wc = blockIdx.z ? Wc1 : Wc0;
    const float* Bc = blockIdx.z ? Bc1 : Bc0;
    float*       XC = blockIdx.z ? XC1 : XC0;
    int t = blockIdx.x * blockDim.x + threadIdx.x;
    if (t >= BB * DIN * LL) return;
    int w = t % WW_;
    int h = (t / WW_) % HH;
    int d = (t / LL) % DIN;
    int b = t / (LL * DIN);
    float acc = Bc[d];
    #pragma unroll
    for (int i = 0; i < 3; ++i) {
        int hh = h + i - 1;
        if (hh < 0 || hh >= HH) continue;
        #pragma unroll
        for (int j = 0; j < 3; ++j) {
            int ww = w + j - 1;
            if (ww < 0 || ww >= WW_) continue;
            acc += Wc[d * 9 + i * 3 + j] * P[((size_t)b * LL + hh * WW_ + ww) * DIN + d];
        }
    }
    float s = acc / (1.f + __expf(-acc));       // SiLU
    XC[((size_t)b * DIN + d) * LL + h * WW_ + w] = s;
}

// ---------------- K3: x_proj, dbl[b,k,c,l] = sum_d xpw[k,c,d] * xs[b,k,d,l] ----------------
__global__ void k_xproj(const float* __restrict__ XC0, const float* __restrict__ XC1,
                        const float* __restrict__ XPW,
                        float* __restrict__ DBL0, float* __restrict__ DBL1)
{
    const float* XC  = blockIdx.z ? XC1 : XC0;
    float*       DBL = blockIdx.z ? DBL1 : DBL0;
    int t = blockIdx.x * blockDim.x + threadIdx.x;
    if (t >= BB * KDIR * LL) return;
    int p = t % LL;
    int k = (t / LL) % KDIR;
    int b = t / (LL * KDIR);
    int pos = scan_pos(k, p);
    float acc[14];
    #pragma unroll
    for (int c = 0; c < 14; ++c) acc[c] = 0.f;
    const float* xc = XC + (size_t)b * DIN * LL + pos;
    const float* wp = XPW + k * 14 * DIN;
    for (int d = 0; d < DIN; ++d) {
        float u = xc[(size_t)d * LL];
        #pragma unroll
        for (int c = 0; c < 14; ++c) acc[c] += wp[c * DIN + d] * u;
    }
    float* o = DBL + (size_t)(b * KDIR + k) * 14 * LL + p;
    #pragma unroll
    for (int c = 0; c < 14; ++c) o[(size_t)c * LL] = acc[c];
}

// ---------------- K4: selective scan. One block per (b,k,modality); thread = channel d. ----
// The 14 wave-uniform streams (dtl x6, B x4 from own modality; C x4 from the other) are
// staged through LDS in double-buffered 128-step chunks: per-step accesses become
// broadcast ds_loads, global traffic becomes bulk coalesced loads overlapped with compute,
// and we pay 1 barrier per 128 steps instead of per-step load-counter waits.
__global__ void k_scan(const float* __restrict__ XC0, const float* __restrict__ XC1,
                       const float* __restrict__ DBL0, const float* __restrict__ DBL1,
                       const float* __restrict__ DTW, const float* __restrict__ DTB,
                       const float* __restrict__ ALOG, const float* __restrict__ DS_,
                       float* __restrict__ YS0, float* __restrict__ YS1)
{
    __shared__ float sd[2][14 * SCH];          // 14 KB

    int z = blockIdx.z;
    const float* XC   = z ? XC1 : XC0;
    const float* DBLs = z ? DBL1 : DBL0;       // own modality: dt (0..5) + B (6..9)
    const float* DBLo = z ? DBL0 : DBL1;       // other modality: C (10..13)
    float* YS = z ? YS1 : YS0;

    int d = threadIdx.x;                       // 0..191
    int k = blockIdx.x % KDIR;
    int b = blockIdx.x / KDIR;

    float dtw[RNK];
    #pragma unroll
    for (int r = 0; r < RNK; ++r) dtw[r] = DTW[(k * DIN + d) * RNK + r];
    float dtb = DTB[k * DIN + d];
    float An[NST];
    #pragma unroll
    for (int n = 0; n < NST; ++n) An[n] = -__expf(ALOG[(k * DIN + d) * NST + n]);
    float Dd = DS_[k * DIN + d];

    const float* ub = XC + ((size_t)b * DIN + d) * LL;
    const float* dS = DBLs + (size_t)(b * KDIR + k) * 14 * LL;
    const float* dO = DBLo + (size_t)(b * KDIR + k) * 14 * LL;
    float* ys = YS + ((size_t)(b * KDIR + k) * DIN + d) * LL;

    // stage chunk 0
    for (int t = threadIdx.x; t < 14 * SCH; t += blockDim.x) {
        int c = t / SCH, i = t % SCH;
        const float* src = (c < 10) ? dS : dO;
        sd[0][t] = src[(size_t)c * LL + i];
    }
    __syncthreads();

    float hst[NST];
    #pragma unroll
    for (int n = 0; n < NST; ++n) hst[n] = 0.f;

    int buf = 0;
    for (int ci = 0; ci < LL / SCH; ++ci) {
        int p0 = ci * SCH;
        // issue next-chunk staging into the other buffer (overlaps with compute below)
        if (p0 + SCH < LL) {
            int p1 = p0 + SCH;
            for (int t = threadIdx.x; t < 14 * SCH; t += blockDim.x) {
                int c = t / SCH, i = t % SCH;
                const float* src = (c < 10) ? dS : dO;
                sd[buf ^ 1][t] = src[(size_t)c * LL + p1 + i];
            }
        }
        const float* sdb = sd[buf];
        for (int i = 0; i < SCH; ++i) {
            int p = p0 + i;
            int pos = scan_pos(k, p);
            float u = ub[pos];
            if (p + 96 < LL) __builtin_prefetch(&ub[scan_pos(k, p + 96)], 0, 1);

            float x = dtb;
            #pragma unroll
            for (int r = 0; r < RNK; ++r) x += dtw[r] * sdb[r * SCH + i];
            float dt = (x > 20.f) ? x : log1pf(__expf(x));   // softplus

            float du = dt * u;
            float y = 0.f;
            #pragma unroll
            for (int n = 0; n < NST; ++n) {
                float Bv = sdb[(RNK + n) * SCH + i];
                float Cv = sdb[(RNK + NST + n) * SCH + i];
                hst[n] = hst[n] * __expf(dt * An[n]) + du * Bv;
                y += hst[n] * Cv;
            }
            y += Dd * u;
            ys[p] = y;
        }
        __syncthreads();            // next-chunk stores complete AND all reads of buf done
        buf ^= 1;
    }
}

// ---------------- K5: cross_merge: 4-direction gather-add ----------------
__global__ void k_merge(const float* __restrict__ YS0, const float* __restrict__ YS1,
                        float* __restrict__ YM0, float* __restrict__ YM1)
{
    const float* YS = blockIdx.z ? YS1 : YS0;
    float*       YM = blockIdx.z ? YM1 : YM0;
    int t = blockIdx.x * blockDim.x + threadIdx.x;
    if (t >= BB * DIN * LL) return;
    int l = t % LL;
    int d = (t / LL) % DIN;
    int b = t / (LL * DIN);
    int h = l / WW_, w = l % WW_;
    int lt = w * HH + h;
    size_t ks = (size_t)DIN * LL;
    size_t base = (size_t)b * KDIR * ks + (size_t)d * LL;
    float y = YS[base + l]
            + YS[base + 1 * ks + lt]
            + YS[base + 2 * ks + (LL - 1 - l)]
            + YS[base + 3 * ks + (LL - 1 - lt)];
    YM[((size_t)b * DIN + d) * LL + l] = y;
}

// ---------------- K6: layernorm over D_inner; emit f16 A-matrix (b,l,d) for out_proj ----------------
__global__ void k_lnorm(const float* __restrict__ YM0, const float* __restrict__ YM1,
                        const float* __restrict__ G0, const float* __restrict__ Bt0,
                        const float* __restrict__ G1, const float* __restrict__ Bt1,
                        _Float16* __restrict__ YL0, _Float16* __restrict__ YL1)
{
    int z = blockIdx.z;
    const float* YM = z ? YM1 : YM0;
    const float* G  = z ? G1 : G0;
    const float* Bb = z ? Bt1 : Bt0;
    _Float16*    YL = z ? YL1 : YL0;
    int t = blockIdx.x * blockDim.x + threadIdx.x;
    if (t >= BB * LL) return;
    int l = t % LL;
    int b = t / LL;
    const float* ym = YM + (size_t)b * DIN * LL + l;
    float s = 0.f, ss = 0.f;
    for (int d = 0; d < DIN; ++d) { float v = ym[(size_t)d * LL]; s += v; ss += v * v; }
    float mu  = s / (float)DIN;
    float var = ss / (float)DIN - mu * mu;
    float inv = rsqrtf(var + 1e-5f);
    _Float16* yl = YL + ((size_t)b * LL + l) * DIN;
    for (int d = 0; d < DIN; ++d) {
        float v = (ym[(size_t)d * LL] - mu) * inv * G[d] + Bb[d];
        yl[d] = (_Float16)v;
    }
}

// ---------------- K7: out_proj WMMA + residual add into d_out ----------------
__global__ void k_gemm_out(const _Float16* __restrict__ A0, const _Float16* __restrict__ A1,
                           const float* __restrict__ W0, const float* __restrict__ W1,
                           const float* __restrict__ R0, const float* __restrict__ R1,
                           float* __restrict__ Y0, float* __restrict__ Y1,
                           int M, int N, int Kd)
{
    const _Float16* A  = blockIdx.z ? A1 : A0;
    const float*    Wt = blockIdx.z ? W1 : W0;
    const float*    Rs = blockIdx.z ? R1 : R0;
    float*          Y  = blockIdx.z ? Y1 : Y0;

    int wave = threadIdx.x >> 5;
    int lane = threadIdx.x & 31;
    int tiles = (M >> 4) * (N >> 4);
    int tile  = blockIdx.x * (blockDim.x >> 5) + wave;
    if (tile >= tiles) return;
    int nt = tile % (N >> 4);
    int mt = tile / (N >> 4);

    int row  = lane & 15;
    int half = lane >> 4;
    int kb8  = half * 8;
    int kb16 = half * 16;

    v8f acc = {0.f, 0.f, 0.f, 0.f, 0.f, 0.f, 0.f, 0.f};
    for (int kk = 0; kk < Kd; kk += 32) {
        v16h a, bm;
        const _Float16* ap = A + (size_t)(mt * 16 + row) * Kd + kk + kb8;
        #pragma unroll
        for (int e = 0; e < 8; ++e) { a[e] = ap[e]; a[8 + e] = ap[16 + e]; }
        const float* bp = Wt + (size_t)(nt * 16 + row) * Kd + kk + kb8;
        #pragma unroll
        for (int e = 0; e < 8; ++e) { bm[e] = (_Float16)bp[e]; bm[8 + e] = (_Float16)bp[16 + e]; }
        acc = __builtin_amdgcn_wmma_f32_16x16x32_f16(false, a, false, bm, (short)0, acc, false, false);
    }
    size_t obase = (size_t)(mt * 16 + half * 8) * N + nt * 16 + row;
    #pragma unroll
    for (int v = 0; v < 8; ++v) Y[obase + (size_t)v * N] = Rs[obase + (size_t)v * N] + acc[v];
}

extern "C" void kernel_launch(void* const* d_in, const int* in_sizes, int n_in,
                              void* d_out, int out_size, void* d_ws, size_t ws_size,
                              hipStream_t stream)
{
    (void)in_sizes; (void)n_in; (void)out_size; (void)ws_size;
    const float* x_rgb = (const float*)d_in[0];
    const float* x_e   = (const float*)d_in[1];
    const float* wpx   = (const float*)d_in[2];
    const float* wpe   = (const float*)d_in[3];
    const float* cwx   = (const float*)d_in[4];
    const float* cbx   = (const float*)d_in[5];
    const float* cwe   = (const float*)d_in[6];
    const float* cbe   = (const float*)d_in[7];
    const float* xpw   = (const float*)d_in[8];
    const float* dtw   = (const float*)d_in[9];
    const float* dtb   = (const float*)d_in[10];
    const float* alog  = (const float*)d_in[11];
    const float* dsv   = (const float*)d_in[12];
    const float* lnrg  = (const float*)d_in[13];
    const float* lnrb  = (const float*)d_in[14];
    const float* lneg  = (const float*)d_in[15];
    const float* lneb  = (const float*)d_in[16];
    const float* wox   = (const float*)d_in[17];
    const float* woe   = (const float*)d_in[18];

    float* ws = (float*)d_ws;
    const size_t S1 = (size_t)BB * LL * DIN;        // 3,145,728 floats
    const size_t SD = (size_t)BB * KDIR * 14 * LL;  //   917,504 floats
    const size_t SY = (size_t)BB * KDIR * DIN * LL; // 12,582,912 floats

    float* PR   = ws;               // in_proj out rgb  (reused later as YMR)
    float* PE   = ws + S1;          // in_proj out e    (reused later as YME)
    float* XCR  = ws + 2 * S1;      // conv out rgb (b,d,l)
    float* XCE  = ws + 3 * S1;      // conv out e
    float* DBLR = ws + 4 * S1;      // x_proj out rgb (b,k,14,l)
    float* DBLE = DBLR + SD;        // x_proj out e
    float* YSR  = DBLE + SD;        // scan out rgb (b,k,d,l)
    float* YSE  = YSR + SY;         // scan out e
    float* YMR  = PR;               // merged rgb (b,d,l)  -- overlays PR (dead after conv)
    float* YME  = PE;
    _Float16* YLR = (_Float16*)YSR; // layernormed f16 (b,l,d) -- overlays YSR (dead after merge)
    _Float16* YLE = YLR + S1;

    float* outR = (float*)d_out;
    float* outE = outR + (size_t)BB * LL * CC;

    // K1: in_proj (M=16384, N=192, K=96), both modalities via grid.z
    {
        int tiles  = (MTOT / 16) * (DIN / 16);          // 12288
        int blocks = (tiles + 7) / 8;                   // 8 waves / 256-thread block
        k_gemm_in<<<dim3(blocks, 1, 2), 256, 0, stream>>>(x_rgb, x_e, wpx, wpe, PR, PE, MTOT, DIN, CC);
    }
    // K2: depthwise conv + SiLU
    {
        int n = BB * DIN * LL;
        k_dwconv<<<dim3((n + 255) / 256, 1, 2), 256, 0, stream>>>(PR, PE, cwx, cwe, cbx, cbe, XCR, XCE);
    }
    // K3: x_proj (14-wide projection per direction)
    {
        int n = BB * KDIR * LL;
        k_xproj<<<dim3((n + 255) / 256, 1, 2), 256, 0, stream>>>(XCR, XCE, xpw, DBLR, DBLE);
    }
    // K4: selective scan; one 192-thread block per (b, k, modality), LDS double-buffered
    k_scan<<<dim3(BB * KDIR, 1, 2), DIN, 0, stream>>>(XCR, XCE, DBLR, DBLE, dtw, dtb, alog, dsv, YSR, YSE);
    // K5: cross merge
    {
        int n = BB * DIN * LL;
        k_merge<<<dim3((n + 255) / 256, 1, 2), 256, 0, stream>>>(YSR, YSE, YMR, YME);
    }
    // K6: layernorm -> f16 A matrix
    {
        int n = BB * LL;
        k_lnorm<<<dim3((n + 255) / 256, 1, 2), 256, 0, stream>>>(YMR, YME, lnrg, lnrb, lneg, lneb, YLR, YLE);
    }
    // K7: out_proj (M=16384, N=96, K=192) + residual
    {
        int tiles  = (MTOT / 16) * (CC / 16);           // 6144
        int blocks = (tiles + 7) / 8;                   // 768
        k_gemm_out<<<dim3(blocks, 1, 2), 256, 0, stream>>>(YLR, YLE, wox, woe, x_rgb, x_e, outR, outE, MTOT, CC, DIN);
    }
}
